// AlltagCopyCtxGenerator_69801808495260
// MI455X (gfx1250) — matrix operand order repaired and checked by reading, compiled, not run on hardware
//
#include <hip/hip_runtime.h>
#include <hip/hip_bf16.h>
#include <math.h>

// ---------------------------------------------------------------------------
// Problem constants (from reference)
// ---------------------------------------------------------------------------
#define BB   32
#define LL   128
#define NN   4096          // B*L tokens
#define HSS  512
#define PP   12
#define MM_  1024
#define VV   50000
#define DD   128
#define CHL  16

// output layout (floats, concatenated in reference return order)
#define OW_   ((size_t)0)          // out_word        4096
#define OC_   ((size_t)4096)       // obf_char        65536
#define OP_   ((size_t)69632)      // psr_out         524288
#define OA_   ((size_t)593920)     // atk_out         524288
#define OMK_  ((size_t)1118208)    // obf_mask        4096
#define OCM_  ((size_t)1122304)    // cpy_mask        4096
#define OPM_  ((size_t)1126400)    // pri_mask        4096
#define OCL_  ((size_t)1130496)    // cpy_loss        1
#define OEL_  ((size_t)1130497)    // ent_loss        1

// LDS strides (pad to dodge bank conflicts; keep 16B row alignment)
#define AST  (HSS + 8)             // A tile stride in halves (1040B = 65*16)
#define LST  (512 + 8)             // logits chunk stride in floats

typedef __attribute__((ext_vector_type(16))) _Float16 v16h;
typedef __attribute__((ext_vector_type(8)))  _Float16 v8h;
typedef __attribute__((ext_vector_type(8)))  float    v8f;

// ---------------------------------------------------------------------------
// Hash-based Gumbel noise (structural stand-in for jax threefry; same
// (key, row, col) dependency pattern, deterministic across calls)
// ---------------------------------------------------------------------------
__device__ __forceinline__ unsigned long long splitmix64(unsigned long long x) {
  x += 0x9E3779B97F4A7C15ull;
  x = (x ^ (x >> 30)) * 0xBF58476D1CE4E5B9ull;
  x = (x ^ (x >> 27)) * 0x94D049BB133111EBull;
  return x ^ (x >> 31);
}

__device__ __forceinline__ float gumbel_noise(unsigned key, unsigned row, unsigned col) {
  unsigned long long h =
      splitmix64(((unsigned long long)key << 42) ^ ((unsigned long long)row << 16) ^ (unsigned long long)col);
  float u = ((float)(h >> 40) + 0.5f) * (1.0f / 16777216.0f);
  u = fminf(fmaxf(u, 1e-6f), 1.0f - 1e-6f);
  return -logf(-logf(u));
}

// ---------------------------------------------------------------------------
// fp32 -> fp16 conversion (grid-stride) — used for ctx
// ---------------------------------------------------------------------------
__global__ void f32_to_f16_kernel(const float* __restrict__ src, _Float16* __restrict__ dst, int n) {
  for (int i = blockIdx.x * blockDim.x + threadIdx.x; i < n; i += gridDim.x * blockDim.x)
    dst[i] = (_Float16)src[i];
}

// ---------------------------------------------------------------------------
// dec_W[p][k][n] (f32, HSxM) -> Wt[p][n][k] (f16, MxHS), 32x32 tiles via LDS
// so each lane's WMMA B-fragment is two 8-half runs contiguous along K.
// ---------------------------------------------------------------------------
__global__ __launch_bounds__(256) void convert_transpose_kernel(const float* __restrict__ src,
                                                                _Float16* __restrict__ dst) {
  __shared__ _Float16 t[32][33];
  const int p  = blockIdx.z;
  const int n0 = blockIdx.x * 32;
  const int k0 = blockIdx.y * 32;
  const int tx = threadIdx.x;       // 0..31
  const int ty = threadIdx.y;       // 0..7
  for (int i = ty; i < 32; i += 8)
    t[i][tx] = (_Float16)src[((size_t)p * HSS + k0 + i) * MM_ + n0 + tx];
  __syncthreads();
  for (int i = ty; i < 32; i += 8)
    dst[((size_t)p * MM_ + n0 + i) * HSS + k0 + tx] = t[tx][i];
}

// ---------------------------------------------------------------------------
// Fused per-(p, 16-row-tile) kernel:
//   logits = ctx16 @ dec_W16[p] + dec_b[p]   (WMMA f32_16x16x32_f16)
//   -> row entropy term  log(s1) - s2/s1     (online-merged over 2 col chunks)
//   -> Gumbel argmax index over M=1024
// Block: 256 threads = 8 waves. ks-outer / 4-N-tile-inner with 4 accumulators,
// A frags from LDS (2x ds_load_b128), B frags from pre-transposed Wt
// (2x global_load_b128). A tile staged with async global->LDS (ASYNCcnt).
// ---------------------------------------------------------------------------
__global__ __launch_bounds__(256) void gemm_softmax_kernel(
    const _Float16* __restrict__ ctx16, const _Float16* __restrict__ wt16,
    const float* __restrict__ bias, float* __restrict__ entRow, int* __restrict__ idxRow) {
  __shared__ _Float16 As[16 * AST];
  __shared__ float    Ls[16 * LST];

  const int rowTile = blockIdx.x;     // 0..255
  const int p       = blockIdx.y;     // 0..11
  const int tid     = threadIdx.x;
  const int lane    = tid & 31;
  const int wave    = tid >> 5;
  const int rowBase = rowTile * 16;

  // --- stage A tile (16 x 512 halves = 16KB) via async global->LDS -------
  // Each thread moves 8 halves (16B) per step; 4 steps cover the tile.
  for (int i = tid * 8; i < 16 * HSS; i += 256 * 8) {
    int r = i >> 9;                    // /512
    int k = i & (HSS - 1);
    unsigned           ldsOff = (unsigned)(unsigned long long)&As[r * AST + k];
    unsigned long long gaddr  = (unsigned long long)(ctx16 + (size_t)(rowBase + r) * HSS + k);
    asm volatile("global_load_async_to_lds_b128 %0, %1, off" ::"v"(ldsOff), "v"(gaddr) : "memory");
  }
  asm volatile("s_wait_asynccnt 0x0" ::: "memory");

  const int m  = lane & 15;   // row within A tile / column within B frag
  const int lh = lane >> 4;   // lane half

  // online softmax running state (reduction role): 16 threads per row
  const int rrow = tid >> 4;
  const int rj   = tid & 15;
  const int gRow = rowBase + rrow;

  float run_mx = -3.0e38f, run_s1 = 0.f, run_s2 = 0.f, run_best = -3.0e38f;
  int   run_bidx = 0;

  for (int chunk = 0; chunk < 2; ++chunk) {
    __syncthreads();   // A staged / previous reduction done

    // ---- compute phase: 4 accumulators, A frag reused 4x per ks ----
    v8f acc0 = {}, acc1 = {}, acc2 = {}, acc3 = {};
    const int nbase0 = (chunk * 32 + wave * 4) * 16;   // this wave's 64 columns
    for (int ks = 0; ks < 16; ++ks) {
      // A frag (16x32 f16): two contiguous 8-half runs per lane from LDS
      const _Float16* Ap = &As[m * AST + ks * 32 + (lh << 3)];
      v8h a0 = *(const v8h*)(Ap);
      v8h a1 = *(const v8h*)(Ap + 16);
      v16h a = __builtin_shufflevector(a0, a1, 0, 1, 2, 3, 4, 5, 6, 7,
                                       8, 9, 10, 11, 12, 13, 14, 15);
      const _Float16* Bbase =
          wt16 + ((size_t)p * MM_ + nbase0 + m) * HSS + ks * 32 + (lh << 3);
      #pragma unroll
      for (int t = 0; t < 4; ++t) {
        // B frag (32x16 f16): two contiguous 8-half runs along K (pre-transposed)
        const _Float16* Bp = Bbase + (size_t)t * 16 * HSS;
        v8h b0 = *(const v8h*)(Bp);
        v8h b1 = *(const v8h*)(Bp + 16);
        v16h b = __builtin_shufflevector(b0, b1, 0, 1, 2, 3, 4, 5, 6, 7,
                                         8, 9, 10, 11, 12, 13, 14, 15);
        __builtin_prefetch(Bp + 32, 0, 0);   // next ks slab of this column group
        v8f cin = (t == 0) ? acc0 : (t == 1) ? acc1 : (t == 2) ? acc2 : acc3;
        v8f cout = __builtin_amdgcn_wmma_f32_16x16x32_f16(
            /*neg_a=*/false, a, /*neg_b=*/false, b,
            /*c_mod=*/(short)0, cin, /*reuse_a=*/false, /*reuse_b=*/false);
        if (t == 0) acc0 = cout; else if (t == 1) acc1 = cout;
        else if (t == 2) acc2 = cout; else acc3 = cout;
      }
    }
    // scatter C (16x16 f32): VGPR r -> row r + 8*lh, lane -> col
    #pragma unroll
    for (int t = 0; t < 4; ++t) {
      const int colLocal = (wave * 4 + t) * 16 + m;
      v8f acc = (t == 0) ? acc0 : (t == 1) ? acc1 : (t == 2) ? acc2 : acc3;
      #pragma unroll
      for (int r = 0; r < 8; ++r)
        Ls[(r + (lh << 3)) * LST + colLocal] = acc[r];
    }
    __syncthreads();

    // ---- reduction phase: 16 threads per row over this 512-col chunk ----
    float mx = -3.0e38f, best = -3.0e38f;
    int   bidx = 0;
    for (int c = rj; c < 512; c += 16) {
      int col = chunk * 512 + c;
      float x = Ls[rrow * LST + c] + bias[p * MM_ + col];
      mx = fmaxf(mx, x);
      float v = x + gumbel_noise((unsigned)p, (unsigned)gRow, (unsigned)col);
      if (v > best) { best = v; bidx = col; }
    }
    #pragma unroll
    for (int o = 8; o > 0; o >>= 1) {
      mx = fmaxf(mx, __shfl_xor(mx, o, 16));
      float ob = __shfl_xor(best, o, 16);
      int   oi = __shfl_xor(bidx, o, 16);
      if (ob > best) { best = ob; bidx = oi; }
    }
    float s1 = 0.f, s2 = 0.f;
    for (int c = rj; c < 512; c += 16) {
      int col = chunk * 512 + c;
      float x  = Ls[rrow * LST + c] + bias[p * MM_ + col];
      float tt = x - mx;
      float ee = __expf(tt);
      s1 += ee;
      s2 += tt * ee;
    }
    #pragma unroll
    for (int o = 8; o > 0; o >>= 1) {
      s1 += __shfl_xor(s1, o, 16);
      s2 += __shfl_xor(s2, o, 16);
    }
    // online merge (guard -inf * 0)
    float nm = fmaxf(run_mx, mx);
    float cA = (run_s1 > 0.f) ? __expf(run_mx - nm) : 0.f;
    float cB = __expf(mx - nm);
    float s2n = ((run_s1 > 0.f) ? (run_s2 + (run_mx - nm) * run_s1) * cA : 0.f) +
                (s2 + (mx - nm) * s1) * cB;
    run_s1 = run_s1 * cA + s1 * cB;
    run_s2 = s2n;
    run_mx = nm;
    if (best > run_best) { run_best = best; run_bidx = bidx; }
  }

  if (rj == 0) {
    // sum_j -logp_j * exp(logp_j) = log(s1) - s2/s1
    entRow[(size_t)p * NN + gRow] = logf(run_s1) - run_s2 / run_s1;
    idxRow[(size_t)p * NN + gRow] = run_bidx;
  }
}

// ---------------------------------------------------------------------------
// Entropy accumulation: sum ent_row over tokens with pos==p, and counts
// ---------------------------------------------------------------------------
__global__ void accum_ent_kernel(const int* __restrict__ posIn, const float* __restrict__ entRow,
                                 float* __restrict__ entSum, int* __restrict__ cntP) {
  int n = blockIdx.x * blockDim.x + threadIdx.x;
  if (n >= NN) return;
  int p = posIn[n];
  if (p >= 0 && p < PP) {
    atomicAdd(&entSum[p], entRow[(size_t)p * NN + n]);
    atomicAdd(&cntP[p], 1);
  }
}

// ---------------------------------------------------------------------------
// Per-token tail: copy MLP, copy-gate Gumbel, gathers, masks, cpy_loss
// One block (64 threads) per token.
// ---------------------------------------------------------------------------
__global__ __launch_bounds__(64) void token_kernel(
    const float* __restrict__ ctx, const float* __restrict__ W1, const float* __restrict__ b1,
    const float* __restrict__ W2, const float* __restrict__ b2,
    const int* __restrict__ wordIn, const int* __restrict__ posIn, const int* __restrict__ maskIn,
    const int* __restrict__ words, const int* __restrict__ lut,
    const float* __restrict__ psrW, const float* __restrict__ atkW,
    const int* __restrict__ idxRow, float* __restrict__ out,
    float* __restrict__ cpyAcc, int* __restrict__ nrCnt) {
  __shared__ float sh[64];
  __shared__ float sc[2];
  __shared__ int   si[3];

  const int n = blockIdx.x;
  const int j = threadIdx.x;
  const float* crow = ctx + (size_t)n * HSS;

  // h = relu(ctx @ W1 + b1), one hidden unit per thread
  float acc = b1[j];
  for (int k = 0; k < HSS; ++k) acc += crow[k] * W1[k * 64 + j];
  sh[j] = fmaxf(acc, 0.f);
  __syncthreads();

  if (j == 0) {
    float z0 = b2[0], z1 = b2[1];
    for (int q = 0; q < 64; ++q) { z0 += sh[q] * W2[q * 2]; z1 += sh[q] * W2[q * 2 + 1]; }
    float mx  = fmaxf(z0, z1);
    float lse = mx + logf(__expf(z0 - mx) + __expf(z1 - mx));
    float l0 = z0 - lse, l1 = z1 - lse;

    int  pos = posIn[n];
    bool pri = (pos >= 0 && pos <= 3);
    float pc0 = pri ? 0.f : l0;
    float pc1 = pri ? 1.f : l1;

    float v0 = pc0 + gumbel_noise(999u, (unsigned)n, 0u);
    float v1 = pc1 + gumbel_noise(999u, (unsigned)n, 1u);
    float c0 = (v0 >= v1) ? 1.f : 0.f;   // argmax picks index 0 on ties
    float c1 = 1.f - c0;

    int  msk     = maskIn[n];
    bool cpyMask = (c0 == 1.f) && (msk != 0);

    int  word    = wordIn[n];
    int  obfWord = word;
    bool obf     = (pos >= 0 && pos < PP);
    if (obf) obfWord = words[pos * MM_ + idxRow[(size_t)pos * NN + n]];
    int outWord = (int)((float)word * c0 + (float)obfWord * c1);

    out[OW_ + n]   = (float)outWord;
    out[OMK_ + n]  = (obf && !cpyMask) ? 1.f : 0.f;
    out[OCM_ + n]  = cpyMask ? 1.f : 0.f;
    out[OPM_ + n]  = pri ? 1.f : 0.f;

    bool noroot = (msk != 0) && ((n % LL) != 0);
    if (noroot) {
      float mx2  = fmaxf(pc0, pc1);
      float lse2 = mx2 + logf(__expf(pc0 - mx2) + __expf(pc1 - mx2));
      atomicAdd(cpyAcc, -(pc1 - lse2));
      atomicAdd(nrCnt, 1);
    }
    sc[0] = c0; sc[1] = c1;
    si[0] = word; si[1] = obfWord; si[2] = outWord;
  }
  __syncthreads();

  const float c0 = sc[0], c1 = sc[1];
  const int word = si[0], obfWord = si[1], outWord = si[2];
  #pragma unroll
  for (int d = j; d < DD; d += 64) {
    out[OP_ + (size_t)n * DD + d] =
        psrW[(size_t)word * DD + d] * c0 + psrW[(size_t)obfWord * DD + d] * c1;
    out[OA_ + (size_t)n * DD + d] =
        atkW[(size_t)word * DD + d] * c0 + atkW[(size_t)obfWord * DD + d] * c1;
  }
  if (j < CHL) out[OC_ + (size_t)n * CHL + j] = (float)lut[(size_t)outWord * CHL + j];
}

// ---------------------------------------------------------------------------
// Accumulator init + final scalar losses
// ---------------------------------------------------------------------------
__global__ void init_kernel(float* entSum, int* cntP, float* cpyAcc, int* nrCnt) {
  int t = threadIdx.x;
  if (t < PP) { entSum[t] = 0.f; cntP[t] = 0; }
  if (t == 0) { cpyAcc[0] = 0.f; nrCnt[0] = 0; }
}

__global__ void finalize_kernel(const float* entSum, const int* cntP,
                                const float* cpyAcc, const int* nrCnt, float* out) {
  if (threadIdx.x == 0 && blockIdx.x == 0) {
    float ent = 0.f;
    for (int p = 0; p < PP; ++p)
      if (cntP[p] > 0) ent += entSum[p] / ((float)cntP[p] * (float)MM_);
    out[OEL_] = -ent;
    int c = nrCnt[0];
    if (c < 1) c = 1;
    out[OCL_] = cpyAcc[0] / (float)c;
  }
}

// ---------------------------------------------------------------------------
// Launch
// ---------------------------------------------------------------------------
extern "C" void kernel_launch(void* const* d_in, const int* in_sizes, int n_in,
                              void* d_out, int out_size, void* d_ws, size_t ws_size,
                              hipStream_t stream) {
  const float* ctx    = (const float*)d_in[0];
  const float* decW   = (const float*)d_in[1];
  const float* decB   = (const float*)d_in[2];
  const float* psrW   = (const float*)d_in[3];
  const float* atkW   = (const float*)d_in[4];
  const float* W1     = (const float*)d_in[5];
  const float* b1     = (const float*)d_in[6];
  const float* W2     = (const float*)d_in[7];
  const float* b2     = (const float*)d_in[8];
  const int*   wordIn = (const int*)d_in[9];
  const int*   posIn  = (const int*)d_in[10];
  const int*   maskIn = (const int*)d_in[11];
  const int*   words  = (const int*)d_in[12];
  const int*   lut    = (const int*)d_in[13];
  float* out = (float*)d_out;

  char* w = (char*)d_ws;
  _Float16* ctx16  = (_Float16*)(w);                                // 4.0 MiB
  _Float16* wt16   = (_Float16*)(w + (size_t)4 * 1024 * 1024);      // 12.0 MiB (transposed)
  float*    entRow = (float*)(w + (size_t)17 * 1024 * 1024);        // P*N f32
  int*      idxRow = (int*)(w + (size_t)17 * 1024 * 1024 + 196608); // P*N i32
  float*    entSum = (float*)(w + (size_t)17 * 1024 * 1024 + 2 * 196608);
  int*      cntP   = (int*)((char*)entSum + 64);
  float*    cpyAcc = (float*)((char*)entSum + 128);
  int*      nrCnt  = (int*)((char*)entSum + 192);

  init_kernel<<<1, 32, 0, stream>>>(entSum, cntP, cpyAcc, nrCnt);
  f32_to_f16_kernel<<<1024, 256, 0, stream>>>(ctx, ctx16, NN * HSS);
  convert_transpose_kernel<<<dim3(MM_ / 32, HSS / 32, PP), dim3(32, 8), 0, stream>>>(decW, wt16);

  gemm_softmax_kernel<<<dim3(NN / 16, PP), 256, 0, stream>>>(ctx16, wt16, decB, entRow, idxRow);

  accum_ent_kernel<<<NN / 256, 256, 0, stream>>>(posIn, entRow, entSum, cntP);

  token_kernel<<<NN, 64, 0, stream>>>(ctx, W1, b1, W2, b2, wordIn, posIn, maskIn,
                                      words, lut, psrW, atkW, idxRow, out, cpyAcc, nrCnt);

  finalize_kernel<<<1, 1, 0, stream>>>(entSum, cntP, cpyAcc, nrCnt, out);
}
